// EnergyToRateConverter_16604343567213
// MI455X (gfx1250) — matrix operand order repaired and checked by reading, not compiled, not run on hardware
//
#include <hip/hip_runtime.h>

// Eyring rate kernel for MI455X (gfx1250, wave32).
// G_from/G_to gathers run on the matrix pipe as one-hot WMMAs
// (v_wmma_f32_16x16x32_f16, exact split-f16 trick); exponentials on the VALU
// (v_exp_f32). Memory-bound: ~738 MB traffic -> ~32 us at 23.3 TB/s.
// All global loads are issued up front so loadcnt covers WMMA/B-build latency.

typedef __attribute__((ext_vector_type(16))) _Float16     v16h;
typedef __attribute__((ext_vector_type(8)))  float        v8f;
typedef __attribute__((ext_vector_type(4)))  float        v4f;
typedef __attribute__((ext_vector_type(8)))  unsigned int v8u;

namespace {
constexpr int    kNStates = 32;
constexpr int    kNTrans  = 48;
constexpr double kKB = 1.380649e-23;
constexpr double kT  = 298.15;
constexpr double kH  = 6.62607015e-34;
constexpr double kR  = 0.008314462618;
// forward = Pref * exp(-(bar - G)/RT) = Pref * exp2(fma(G, C, -C*bar))
constexpr float kPref = (float)(kKB * kT / kH);
constexpr float kCExp = (float)(1.4426950408889634 / (kR * kT));  // log2(e)/RT

// One-hot 32x16 f16 B-fragment, built per-dword (8 cmp+sel instead of 16).
// Lane layout: column n = lane&15; lane<16 holds K=0..7,16..23, lane>=16 K=8..15,24..31;
// element s holds K = 8*half + s (s<8), 16 + 8*half + (s-8) (s>=8).
__device__ inline v16h onehot_frag(int idx, int laneHalf) {
  const unsigned q   = ((unsigned)idx < 32u && (((unsigned)idx >> 3) & 1u) == (unsigned)laneHalf)
                           ? (((unsigned)idx & 7u) | (((unsigned)idx >> 4) << 3)) : 16u;
  const unsigned sel = 0x3C00u << ((q & 1u) << 4);        // f16 1.0 in low/high half
  const unsigned hot = q >> 1;                            // hot dword (>=8 -> none)
  v8u w;
#pragma unroll
  for (unsigned v = 0; v < 8; ++v) w[v] = (hot == v) ? sel : 0u;
  return __builtin_bit_cast(v16h, w);
}
}

__global__ __launch_bounds__(256) void eyring_wmma_kernel(
    const float* __restrict__ state,              // [B,32]
    const float* __restrict__ barrier,            // [B,48]
    const int*   __restrict__ from_idx,           // [48]
    const int*   __restrict__ to_idx,             // [48]
    const unsigned char* __restrict__ reversible, // [48] (jax bool = 1 byte)
    float* __restrict__ out_fwd,                  // [B,48]
    float* __restrict__ out_rev)                  // [B,48]
{
  const int lane     = (int)(threadIdx.x & 31u);
  const int r        = lane & 15;   // A row-in-tile; B/C column
  const int laneHalf = lane >> 4;
  const long long tileBase =
      (((long long)blockIdx.x * 8) + (long long)(threadIdx.x >> 5)) * 16;

  // ---------------- Issue ALL global loads up front (independent). ---------
  // A source: this lane's 16 f32 of the state tile.
  // ISA 16-bit A layout (16x32): lane<16 holds K=0..7,16..23 of row=lane;
  // lane>=16 holds K=8..15,24..31 of row=lane-16.
  const float* rowp = state + (tileBase + r) * kNStates;
  const v4f x0 = *(const v4f*)(rowp +      8 * laneHalf);
  const v4f x1 = *(const v4f*)(rowp +  4 + 8 * laneHalf);
  const v4f x2 = *(const v4f*)(rowp + 16 + 8 * laneHalf);
  const v4f x3 = *(const v4f*)(rowp + 20 + 8 * laneHalf);

  int fi[3], ti[3];
  unsigned char rvb[3];
#pragma unroll
  for (int nt = 0; nt < 3; ++nt) {
    fi[nt]  = from_idx[nt * 16 + r];
    ti[nt]  = to_idx[nt * 16 + r];
    rvb[nt] = reversible[nt * 16 + r];
  }

  // Barrier values for all 24 outputs this lane owns (C layout:
  // VGPR j -> row tileBase + j + 8*laneHalf, column nt*16 + r).
  const long long rowBase = (tileBase + 8 * laneHalf) * (long long)kNTrans + r;
  float bar[3][8];
#pragma unroll
  for (int nt = 0; nt < 3; ++nt)
#pragma unroll
    for (int j = 0; j < 8; ++j)
      bar[nt][j] = __builtin_nontemporal_load(barrier + rowBase +
                                              j * kNTrans + nt * 16);

  // ---------------- A fragments: exact hi/lo f16 split of f32 state. ------
  float xs[16];
#pragma unroll
  for (int q = 0; q < 4; ++q) {
    xs[q] = x0[q]; xs[4 + q] = x1[q]; xs[8 + q] = x2[q]; xs[12 + q] = x3[q];
  }
  v16h a_hi, a_lo;
#pragma unroll
  for (int e = 0; e < 16; ++e) {
    const float    x = xs[e];
    const _Float16 h = (_Float16)x;
    a_hi[e] = h;
    a_lo[e] = (_Float16)(x - (float)h);   // exact residual; A = hi + lo
  }

  const v8f zero = {};

#pragma unroll
  for (int nt = 0; nt < 3; ++nt) {        // 48 transitions = 3 tiles of 16
    const v16h bf = onehot_frag(fi[nt], laneHalf);
    const v16h bt = onehot_frag(ti[nt], laneHalf);

    // G = A_lo x B + A_hi x B (f32 accumulate) == exact f32 gather.
    v8f gf = __builtin_amdgcn_wmma_f32_16x16x32_f16(false, a_lo, false, bf,
                                                    (short)0, zero, false, false);
    gf     = __builtin_amdgcn_wmma_f32_16x16x32_f16(false, a_hi, false, bf,
                                                    (short)0, gf, false, false);
    v8f gt = __builtin_amdgcn_wmma_f32_16x16x32_f16(false, a_lo, false, bt,
                                                    (short)0, zero, false, false);
    gt     = __builtin_amdgcn_wmma_f32_16x16x32_f16(false, a_hi, false, bt,
                                                    (short)0, gt, false, false);

    const bool rv = rvb[nt] != 0;

    float fwd[8], rev[8];
#pragma unroll
    for (int j = 0; j < 8; ++j) {
      const float nb = bar[nt][j] * (-kCExp);                 // shared term
      fwd[j] = kPref * __builtin_amdgcn_exp2f(__builtin_fmaf(gf[j], kCExp, nb));
      const float rr =
          kPref * __builtin_amdgcn_exp2f(__builtin_fmaf(gt[j], kCExp, nb));
      rev[j] = rv ? rr : 0.0f;
    }

#pragma unroll
    for (int j = 0; j < 8; ++j) {
      const long long off = rowBase + j * kNTrans + nt * 16;
      __builtin_nontemporal_store(fwd[j], out_fwd + off);
      __builtin_nontemporal_store(rev[j], out_rev + off);
    }
  }
}

// Scalar tail for batch counts not divisible by 128 (keeps the WMMA kernel's
// EXEC all-ones requirement trivially satisfied).
__global__ void eyring_tail_kernel(
    const float* __restrict__ state, const float* __restrict__ barrier,
    const int* __restrict__ from_idx, const int* __restrict__ to_idx,
    const unsigned char* __restrict__ reversible,
    float* __restrict__ out_fwd, float* __restrict__ out_rev,
    long long bStart, long long bEnd)
{
  const long long idx   = (long long)blockIdx.x * blockDim.x + threadIdx.x;
  const long long total = (bEnd - bStart) * kNTrans;
  if (idx >= total) return;
  const long long b = bStart + idx / kNTrans;
  const int       t = (int)(idx % kNTrans);
  const float gfv = state[b * kNStates + from_idx[t]];
  const float gtv = state[b * kNStates + to_idx[t]];
  const long long off = b * kNTrans + t;
  const float nb = barrier[off] * (-kCExp);
  out_fwd[off] = kPref * __builtin_amdgcn_exp2f(__builtin_fmaf(gfv, kCExp, nb));
  out_rev[off] = reversible[t]
      ? kPref * __builtin_amdgcn_exp2f(__builtin_fmaf(gtv, kCExp, nb))
      : 0.0f;
}

extern "C" void kernel_launch(void* const* d_in, const int* in_sizes, int n_in,
                              void* d_out, int out_size, void* d_ws, size_t ws_size,
                              hipStream_t stream) {
  const float*         state = (const float*)d_in[0];
  const float*         bar   = (const float*)d_in[1];
  const int*           fi    = (const int*)d_in[2];
  const int*           ti    = (const int*)d_in[3];
  const unsigned char* rev   = (const unsigned char*)d_in[4];

  float* out_fwd = (float*)d_out;
  const long long Bn = (long long)in_sizes[0] / kNStates;
  float* out_rev = out_fwd + Bn * kNTrans;

  const long long blocks = Bn / 128;          // 8 waves x 16 rows per block
  if (blocks > 0) {
    eyring_wmma_kernel<<<(unsigned)blocks, 256, 0, stream>>>(
        state, bar, fi, ti, rev, out_fwd, out_rev);
  }
  const long long rem = Bn % 128;
  if (rem > 0) {
    const long long bStart = Bn - rem;
    const long long n      = rem * kNTrans;
    eyring_tail_kernel<<<(unsigned)((n + 255) / 256), 256, 0, stream>>>(
        state, bar, fi, ti, rev, out_fwd, out_rev, bStart, Bn);
  }
}
